// Covpool_44865228374072
// MI455X (gfx1250) — matrix operand order repaired and checked by reading, not compiled, not run on hardware
//
#include <hip/hip_runtime.h>

typedef __attribute__((ext_vector_type(16))) __bf16 v16bf;
typedef __attribute__((ext_vector_type(8)))  __bf16 v8bf;
typedef __attribute__((ext_vector_type(8)))  float  v8f;
typedef __attribute__((ext_vector_type(4)))  float  v4f;

#define BATCH 64
#define CHAN  128
#define MDIM  4096
#define KT    64      // K-slab width (two WMMA k-steps of 32)
#define LS    72      // padded LDS row stride in bf16 elems (144B -> conflict-free)

union Frag { v16bf v; v8bf h[2]; };

__global__ __launch_bounds__(512)
void covpool_gram_bf16x3(const float* __restrict__ x, float* __restrict__ out)
{
  __shared__ __bf16 sHi[CHAN * LS];
  __shared__ __bf16 sLo[CHAN * LS];
  __shared__ float  sSum[512];
  __shared__ float  sMean[CHAN];

  const int tid  = threadIdx.x;
  const int lane = tid & 31;
  const int wave = tid >> 5;
  const int rb   = blockIdx.x;   // row block: rows [rb*64, rb*64+64)
  const int b    = blockIdx.y;   // batch

  // loader mapping: 512 threads cover 128 channels x 64 floats per slab
  const int lrow = tid >> 2;     // channel 0..127
  const int lq   = tid & 3;      // 16-float quarter of the slab row

  // wmma tile mapping: wave -> (tile row, pair of col tiles)
  const int tr = wave >> 2;                // 0..3
  const int R  = rb * 4 + tr;              // global row tile, rows [16R,16R+16)
  const int cq = wave & 3;                 // col quarter: tiles at 32*cq + {0,16}

  const int sel = lane >> 4;               // K-half selector per ISA layout
  const int mr  = lane & 15;               // row/col within 16x16 tile

  v8f acc[2];
  #pragma unroll
  for (int t = 0; t < 2; ++t)
    #pragma unroll
    for (int i = 0; i < 8; ++i) acc[t][i] = 0.0f;

  const float* __restrict__ xrow =
      x + ((size_t)(b * CHAN + lrow)) * MDIM + lq * 16;

  float csum = 0.0f;

  for (int s = 0; s < MDIM / KT; ++s) {
    const int k0 = s * KT;

    // ---- stream 16 f32/thread, split into bf16 hi/lo, track channel sum ----
    v4f f0 = *(const v4f*)(xrow + k0 + 0);
    v4f f1 = *(const v4f*)(xrow + k0 + 4);
    v4f f2 = *(const v4f*)(xrow + k0 + 8);
    v4f f3 = *(const v4f*)(xrow + k0 + 12);
    if (s + 1 < MDIM / KT) __builtin_prefetch(xrow + k0 + KT, 0, 1);

    float vals[16];
    *(v4f*)&vals[0]  = f0; *(v4f*)&vals[4]  = f1;
    *(v4f*)&vals[8]  = f2; *(v4f*)&vals[12] = f3;

    v8bf h0, h1, l0, l1;
    #pragma unroll
    for (int i = 0; i < 8; ++i) {
      float a  = vals[i];
      float bb = vals[i + 8];
      csum += a + bb;
      __bf16 ah = (__bf16)a;
      __bf16 bh = (__bf16)bb;
      h0[i] = ah;
      h1[i] = bh;
      l0[i] = (__bf16)(a  - (float)ah);
      l1[i] = (__bf16)(bb - (float)bh);
    }

    __syncthreads();                 // previous slab fully consumed
    {
      const int base = lrow * LS + lq * 16;
      *(v8bf*)&sHi[base + 0] = h0;
      *(v8bf*)&sHi[base + 8] = h1;
      *(v8bf*)&sLo[base + 0] = l0;
      *(v8bf*)&sLo[base + 8] = l1;
    }
    __syncthreads();                 // slab visible to all waves

    // ---- two K=32 steps over the slab; bf16x3 product into f32 acc ----
    #pragma unroll
    for (int kk = 0; kk < KT; kk += 32) {
      Frag aHi, aLo;
      const int abase = (16 * R + mr) * LS + kk;
      aHi.h[0] = *(const v8bf*)&sHi[abase + 8 * sel];
      aHi.h[1] = *(const v8bf*)&sHi[abase + 16 + 8 * sel];
      aLo.h[0] = *(const v8bf*)&sLo[abase + 8 * sel];
      aLo.h[1] = *(const v8bf*)&sLo[abase + 16 + 8 * sel];

      #pragma unroll
      for (int t = 0; t < 2; ++t) {
        const int nrow  = 32 * cq + 16 * t + mr;
        const int bbase = nrow * LS + kk + 8 * sel;
        Frag bHi, bLo;
        bHi.h[0] = *(const v8bf*)&sHi[bbase + 0];
        bHi.h[1] = *(const v8bf*)&sHi[bbase + 16];
        bLo.h[0] = *(const v8bf*)&sLo[bbase + 0];
        bLo.h[1] = *(const v8bf*)&sLo[bbase + 16];

        acc[t] = __builtin_amdgcn_wmma_f32_16x16x32_bf16(
            false, aHi.v, false, bHi.v, (short)0, acc[t], false, false);
        acc[t] = __builtin_amdgcn_wmma_f32_16x16x32_bf16(
            false, aHi.v, false, bLo.v, (short)0, acc[t], false, false);
        acc[t] = __builtin_amdgcn_wmma_f32_16x16x32_bf16(
            false, aLo.v, false, bHi.v, (short)0, acc[t], false, false);
      }
    }
  }

  // ---- per-channel means from streamed partial sums ----
  sSum[tid] = csum;
  __syncthreads();
  if (tid < CHAN) {
    sMean[tid] = (sSum[4 * tid] + sSum[4 * tid + 1] +
                  sSum[4 * tid + 2] + sSum[4 * tid + 3]) * (1.0f / (float)MDIM);
  }
  __syncthreads();

  // ---- epilogue: y = G/M - mu mu^T  (C/D layout: lane=N, vgpr v -> M=8*sel+v)
  const float invM = 1.0f / (float)MDIM;
  float* outb = out + (size_t)b * CHAN * CHAN;
  #pragma unroll
  for (int t = 0; t < 2; ++t) {
    const int   gcol = 32 * cq + 16 * t + mr;
    const float muc  = sMean[gcol];
    #pragma unroll
    for (int v = 0; v < 8; ++v) {
      const int grow = 16 * R + 8 * sel + v;
      outb[(size_t)grow * CHAN + gcol] = acc[t][v] * invM - sMean[grow] * muc;
    }
  }
}

extern "C" void kernel_launch(void* const* d_in, const int* in_sizes, int n_in,
                              void* d_out, int out_size, void* d_ws, size_t ws_size,
                              hipStream_t stream) {
  (void)in_sizes; (void)n_in; (void)d_ws; (void)ws_size; (void)out_size;
  const float* x   = (const float*)d_in[0];
  float*       out = (float*)d_out;
  covpool_gram_bf16x3<<<dim3(2, BATCH), dim3(512), 0, stream>>>(x, out);
}
